// LSTM_22557168239098
// MI455X (gfx1250) — compile-verified, matrix operand order checked by reading, not run on hardware
//
#include <hip/hip_runtime.h>
#include <hip/hip_bf16.h>
#include <stdint.h>

#define B_  64
#define S_  512
#define I_  1024
#define H_  1024
#define G4  4096   // 4*H
#define NBLK 64    // persistent blocks: 16 hidden cols each

typedef __attribute__((ext_vector_type(16))) __bf16 v16bf;
typedef __attribute__((ext_vector_type(2)))  __bf16 v2bf;
typedef __attribute__((ext_vector_type(8)))  float  v8f;

union FragBF { v16bf v; uint32_t u[8]; };

// ---- f32 -> bf16 pair, prefer hardware v_cvt_pk_bf16_f32 ----
__device__ inline uint32_t pack2_bf16(float lo, float hi) {
#if defined(__has_builtin) && __has_builtin(__builtin_amdgcn_cvt_pk_bf16_f32)
  union { v2bf v; uint32_t u; } c;
  c.v = __builtin_amdgcn_cvt_pk_bf16_f32(lo, hi);
  return c.u;
#else
  union { float f; uint32_t u; } a, b;
  a.f = lo; b.f = hi;
  uint32_t ua = a.u + 0x7FFFu + ((a.u >> 16) & 1u);
  uint32_t ub = b.u + 0x7FFFu + ((b.u >> 16) & 1u);
  return (ua >> 16) | (ub & 0xFFFF0000u);
#endif
}
__device__ inline uint16_t f32_to_bf16(float x) {
  union { float f; uint32_t u; } a; a.f = x;
  return (uint16_t)((a.u + 0x7FFFu + ((a.u >> 16) & 1u)) >> 16);
}
__device__ inline float bf16_to_f32(uint16_t h) {
  union { uint32_t u; float f; } a; a.u = ((uint32_t)h) << 16;
  return a.f;
}

// ---- bf16 fragment loader for v_wmma_f32_16x16x32_bf16, TN layout ----
// lane 0-15: row = lane,  K = {0..7} u {16..23};  lane 16-31: row = lane-16,
// K = {8..15} u {24..31}  (ISA 7.12.2). Works for global or LDS base pointers.
__device__ inline FragBF load_frag_bf16(const uint32_t* base_u32,
                                        int ld_elems, int row0, int k0, int lane) {
  int m  = lane & 15;
  int kb = (lane >> 4) << 3;
  size_t e = (size_t)(row0 + m) * ld_elems + k0 + kb;   // bf16-elem index (even, 16B aligned)
  uint4 a = *(const uint4*)(base_u32 + (e >> 1));
  uint4 b = *(const uint4*)(base_u32 + ((e + 16) >> 1));
  FragBF r;
  r.u[0] = a.x; r.u[1] = a.y; r.u[2] = a.z; r.u[3] = a.w;
  r.u[4] = b.x; r.u[5] = b.y; r.u[6] = b.z; r.u[7] = b.w;
  return r;
}

__device__ inline v8f wmma_bf16(const FragBF& a, const FragBF& b, v8f c) {
  return __builtin_amdgcn_wmma_f32_16x16x32_bf16(
      false, a.v, false, b.v, (short)0, c, false, false);
}

// ---------------------------------------------------------------------------
// Kernel 0: zero h double-buffer (bf16), c state (f32), grid-barrier counter
// ---------------------------------------------------------------------------
__global__ void lstm_init_kernel(uint32_t* __restrict__ hbufs, float* __restrict__ cbuf,
                                 int* __restrict__ counter) {
  int i = blockIdx.x * blockDim.x + threadIdx.x;
  if (i < 2 * B_ * H_ / 2) hbufs[i] = 0u;
  if (i < B_ * H_)         cbuf[i]  = 0.f;
  if (i == 0)              *counter = 0;
}

// ---------------------------------------------------------------------------
// Kernel 1: generic f32 -> packed bf16 conversion (8 floats / thread).
// ---------------------------------------------------------------------------
__global__ void cvt_f32_bf16_kernel(const float* __restrict__ src,
                                    uint32_t* __restrict__ dst, int n8) {
  int i = blockIdx.x * blockDim.x + threadIdx.x;
  if (i < n8) {
    const float4* s = (const float4*)src + 2 * (size_t)i;
    float4 f0 = s[0], f1 = s[1];
    uint4 o;
    o.x = pack2_bf16(f0.x, f0.y);
    o.y = pack2_bf16(f0.z, f0.w);
    o.z = pack2_bf16(f1.x, f1.y);
    o.w = pack2_bf16(f1.z, f1.w);
    ((uint4*)dst)[i] = o;
  }
}

// ---------------------------------------------------------------------------
// Kernel 2: xg = x @ W_ih^T + (b_ih + b_hh), stored bf16.  [32768 x 4096], K=1024
// 8 waves / block; each wave: 32x64 strip (2 M-tiles x 4 N-tiles, 8 wmma/K-chunk)
// ---------------------------------------------------------------------------
__global__ void __launch_bounds__(256)
lstm_xg_gemm_kernel(const uint32_t* __restrict__ xbf, const uint32_t* __restrict__ wihbf,
                    const float* __restrict__ bih, const float* __restrict__ bhh,
                    uint16_t* __restrict__ xg) {
  int lane   = threadIdx.x & 31;
  int gw     = blockIdx.x * 8 + (threadIdx.x >> 5);
  int mtile  = gw >> 6;
  int nstrip = gw & 63;
  int row0 = mtile << 5;
  int col0 = nstrip << 6;

  v8f acc0[4] = {}, acc1[4] = {};
  for (int k0 = 0; k0 < I_; k0 += 32) {
    if (k0 + 32 < I_) {
      __builtin_prefetch(xbf + (((size_t)row0 * I_ + k0 + 32 + (lane & 15) * I_) >> 1), 0, 0);
    }
    FragBF a0 = load_frag_bf16(xbf, I_, row0,      k0, lane);
    FragBF a1 = load_frag_bf16(xbf, I_, row0 + 16, k0, lane);
#pragma unroll
    for (int j = 0; j < 4; ++j) {
      FragBF bf = load_frag_bf16(wihbf, I_, col0 + 16 * j, k0, lane);
      acc0[j] = wmma_bf16(a0, bf, acc0[j]);
      acc1[j] = wmma_bf16(a1, bf, acc1[j]);
    }
  }
  int n  = lane & 15;
  int m0 = (lane >> 4) << 3;
#pragma unroll
  for (int j = 0; j < 4; ++j) {
    int g = col0 + 16 * j + n;
    float bias = bih[g] + bhh[g];
#pragma unroll
    for (int r = 0; r < 8; ++r) {
      xg[(size_t)(row0 + m0 + r) * G4 + g]      = f32_to_bf16(acc0[j][r] + bias);
      xg[(size_t)(row0 + 16 + m0 + r) * G4 + g] = f32_to_bf16(acc1[j][r] + bias);
    }
  }
}

// ---------------------------------------------------------------------------
// Kernel 3: PERSISTENT recurrence. 64 blocks x 512 threads (16 waves), all 512
// timesteps in one launch. Block owns hidden slice [n0, n0+16): its W_hh
// working set (64 rows x 1024 x bf16 = 128 KB) lives in LDS for the whole
// kernel; gates staged in LDS; device-scope atomic grid barrier between steps.
// ---------------------------------------------------------------------------
__global__ void __launch_bounds__(512)
lstm_persistent_kernel(const uint32_t* __restrict__ whh_bf,
                       uint16_t* __restrict__ h0buf, uint16_t* __restrict__ h1buf,
                       float* __restrict__ c_buf,
                       const uint16_t* __restrict__ xg,
                       float* __restrict__ outH, float* __restrict__ outC,
                       int* __restrict__ counter) {
  extern __shared__ char smem[];
  uint32_t* w_lds = (uint32_t*)smem;                    // 64 rows x 512 u32 = 128 KiB
  float*    stage = (float*)(smem + 64 * 512 * 4);      // 64 x STG gate staging
  const int STG = 68;

  int tid   = threadIdx.x;
  int lane  = tid & 31;
  int wid   = tid >> 5;        // 0..15
  int mtile = wid & 3;         // batch rows 16*mtile
  int gsel  = wid >> 2;        // gate 0..3 (i,f,g,o)
  int n0    = blockIdx.x << 4; // hidden slice base (16 cols)

  // ---- one-time: W_hh slice -> LDS. local row r = gsel*16 + nloc, 1024 bf16/row
  for (int i = tid; i < 64 * 128; i += 512) {           // 8192 uint4 granules
    int r = i >> 7, q = i & 127;
    int grow = (r >> 4) * H_ + n0 + (r & 15);
    uint4 v = *(const uint4*)(whh_bf + (((size_t)grow * H_) >> 1) + q * 4);
    *(uint4*)(w_lds + r * 512 + q * 4) = v;
  }
  __syncthreads();

  for (int t = 0; t < S_; ++t) {
    const uint16_t* hin  = (t & 1) ? h1buf : h0buf;
    uint16_t*       hout = (t & 1) ? h0buf : h1buf;
    const uint32_t* hin_u32 = (const uint32_t*)hin;

    // ---- GEMM: one 16x16 tile per wave, A from global h, B from LDS W
    v8f acc = {};
    for (int k0 = 0; k0 < H_; k0 += 32) {
      FragBF af = load_frag_bf16(hin_u32, H_, mtile * 16, k0, lane);
      FragBF bf = load_frag_bf16(w_lds,   H_, gsel * 16,  k0, lane);
      acc = wmma_bf16(af, bf, acc);
    }
    int n  = lane & 15;
    int m0 = (lane >> 4) << 3;
#pragma unroll
    for (int r = 0; r < 8; ++r)
      stage[(mtile * 16 + m0 + r) * STG + gsel * 16 + n] = acc[r];
    __syncthreads();

    // ---- fused LSTM cell update: 64 batch x 16 hidden = 1024 elems
    for (int e = tid; e < 64 * 16; e += 512) {
      int b  = e >> 4;
      int j  = e & 15;
      int nn = n0 + j;
      size_t xrow = ((size_t)b * S_ + t) * G4;
      float gi = stage[b * STG +      j] + bf16_to_f32(xg[xrow + 0 * H_ + nn]);
      float gf = stage[b * STG + 16 + j] + bf16_to_f32(xg[xrow + 1 * H_ + nn]);
      float gg = stage[b * STG + 32 + j] + bf16_to_f32(xg[xrow + 2 * H_ + nn]);
      float go = stage[b * STG + 48 + j] + bf16_to_f32(xg[xrow + 3 * H_ + nn]);
      float i_ = 1.f / (1.f + __expf(-gi));
      float f_ = 1.f / (1.f + __expf(-gf));
      float g_ = tanhf(gg);
      float o_ = 1.f / (1.f + __expf(-go));
      float c_prev = c_buf[b * H_ + nn];
      float c_new  = f_ * c_prev + i_ * g_;
      float h_new  = o_ * tanhf(c_new);
      c_buf[b * H_ + nn] = c_new;
      size_t oidx = ((size_t)b * S_ + t) * H_ + nn;
      outH[oidx] = h_new;
      outC[oidx] = c_new;
      hout[b * H_ + nn] = f32_to_bf16(h_new);
    }

    // ---- device-scope grid barrier (all 64 blocks resident by construction)
    __syncthreads();
    if (tid == 0) {
      __threadfence();
      __hip_atomic_fetch_add(counter, 1, __ATOMIC_RELEASE, __HIP_MEMORY_SCOPE_AGENT);
      int target = (t + 1) * NBLK;
      while (__hip_atomic_load(counter, __ATOMIC_ACQUIRE, __HIP_MEMORY_SCOPE_AGENT) < target)
        __builtin_amdgcn_s_sleep(2);
    }
    __syncthreads();
  }
}

// ---------------------------------------------------------------------------
extern "C" void kernel_launch(void* const* d_in, const int* in_sizes, int n_in,
                              void* d_out, int out_size, void* d_ws, size_t ws_size,
                              hipStream_t stream) {
  const float* x   = (const float*)d_in[0];   // [B,S,I]
  const float* Wih = (const float*)d_in[1];   // [4H,I]
  const float* Whh = (const float*)d_in[2];   // [4H,H]
  const float* bih = (const float*)d_in[3];   // [4H]
  const float* bhh = (const float*)d_in[4];   // [4H]

  char* ws = (char*)d_ws;
  const size_t MB = 1024u * 1024u;
  uint32_t* whhbf = (uint32_t*)(ws);                 //  8 MiB
  uint32_t* wihbf = (uint32_t*)(ws + 8 * MB);        //  8 MiB
  uint32_t* xbf   = (uint32_t*)(ws + 16 * MB);       // 64 MiB
  uint16_t* h0    = (uint16_t*)(ws + 80 * MB);       // 128 KiB
  uint16_t* h1    = (uint16_t*)(ws + 80 * MB + 128 * 1024);
  float*    cbuf  = (float*)   (ws + 80 * MB + 256 * 1024);  // 256 KiB
  int*      cnt   = (int*)     (ws + 80 * MB + 384 * 1024);
  uint16_t* xg    = (uint16_t*)(ws + 80 * MB + 512 * 1024);  // 256 MiB

  float* outH = (float*)d_out;
  float* outC = outH + (size_t)B_ * S_ * H_;

  lstm_init_kernel<<<256, 256, 0, stream>>>((uint32_t*)h0, cbuf, cnt);

  cvt_f32_bf16_kernel<<<(B_ * S_ * I_ / 8 + 255) / 256, 256, 0, stream>>>(x,   xbf,   B_ * S_ * I_ / 8);
  cvt_f32_bf16_kernel<<<(G4 * I_ / 8     + 255) / 256, 256, 0, stream>>>(Wih, wihbf, G4 * I_ / 8);
  cvt_f32_bf16_kernel<<<(G4 * H_ / 8     + 255) / 256, 256, 0, stream>>>(Whh, whhbf, G4 * H_ / 8);

  lstm_xg_gemm_kernel<<<8192, 256, 0, stream>>>(xbf, wihbf, bih, bhh, xg);

  const int lds_bytes = 64 * 512 * 4 + 64 * 68 * 4;   // 128 KiB W + 17 KiB stage
  lstm_persistent_kernel<<<NBLK, 512, lds_bytes, stream>>>(whhbf, h0, h1, cbuf, xg,
                                                           outH, outC, cnt);
}
